// DelayExpansionLayer_1683627180542
// MI455X (gfx1250) — compile-verified, matrix operand order checked by reading, not run on hardware
//
#include <hip/hip_runtime.h>

// Problem geometry (fixed by the reference's setup_inputs):
//   layer_output: (B=64, C=256, H=56, W=56) fp32 -> 205.5 MB streamed once.
//   Memory-bound: 205.5 MB / 23.3 TB/s ~= 8.8 us floor. Stage 1 is pure
//   coalesced streaming; stage 2 (tiny) uses V_WMMA_F32_16X16X4_F32 to do the
//   per-channel row-sum of batch partials on the matrix unit.
#define C_CHANNELS 256
#define HW 3136     // 56*56
#define HW4 784     // HW/4 float4 chunks per (b,c) slice

typedef __attribute__((ext_vector_type(2)))  float    v2f;
typedef __attribute__((ext_vector_type(4)))  float    f4;
typedef __attribute__((ext_vector_type(8)))  float    v8f;
typedef __attribute__((ext_vector_type(16))) _Float16 v16h;

// ---------------------------------------------------------------------------
// Stage 1: one block per (b,c) contiguous 3136-float chunk.
// float4 (global_load_b128) non-temporal streaming loads + deterministic
// fixed-order LDS tree reduction. Writes partials laid out [C][nB].
// ---------------------------------------------------------------------------
__global__ __launch_bounds__(256)
void delay_partial_sums_kernel(const float* __restrict__ x,
                               float* __restrict__ partials, int nB) {
    __shared__ float s[256];
    const int bc = blockIdx.x;           // bc = b*C + c (input memory order)
    const int b  = bc / C_CHANNELS;
    const int c  = bc % C_CHANNELS;
    const f4* p = (const f4*)(x + (size_t)bc * HW);   // 16B aligned (12544B chunks)

    float acc = 0.0f;
    for (int i = threadIdx.x; i < HW4; i += 256) {
        f4 v = __builtin_nontemporal_load(&p[i]);     // stream: no L2 reuse
        acc += (v.x + v.y) + (v.z + v.w);
    }
    s[threadIdx.x] = acc;
    __syncthreads();
    for (int st = 128; st > 0; st >>= 1) {            // fixed-order tree
        if (threadIdx.x < st) s[threadIdx.x] += s[threadIdx.x + st];
        __syncthreads();
    }
    if (threadIdx.x == 0) partials[c * nB + b] = s[0];
}

// ---------------------------------------------------------------------------
// Stage 2: single block, 512 threads = 16 waves (wave32). Each wave reduces a
// 16-channel group: rowsum(P[16 x nB]) = WMMA(A = 16x4 f32 tile, B = ones).
// Then mean -> round(1e-6) -> searchsorted nearest lookup -> max -> broadcast.
// ---------------------------------------------------------------------------
__global__ __launch_bounds__(512)
void delay_finalize_kernel(const float* __restrict__ partials,
                           const float* __restrict__ keys,
                           const float* __restrict__ vals,
                           const int*   __restrict__ icp,
                           const int*   __restrict__ ocp,
                           float* __restrict__ out,
                           int nB, int K, int outN) {
    __shared__ float s_sum[C_CHANNELS];
    __shared__ float s_val[C_CHANNELS];

    const int tid     = threadIdx.x;
    const int lane    = tid & 31;
    const int wave    = tid >> 5;
    const int ch_base = wave * 16;       // 16 waves * 16 = 256 channels

    // A-matrix row for this lane: 16x4 (and 16x32 f16) layouts both map
    // M = lane & 15 in each half-wave; `sel` picks which k-slice the lane owns.
    const int row = lane & 15;
    const int sel = lane >> 4;           // 0: lanes 0-15, 1: lanes 16-31
    v8f acc = {};                        // f32 16x16 accumulator (8 VGPRs)

#if __has_builtin(__builtin_amdgcn_wmma_f32_16x16x4_f32)
    // Exact fp32 path: D = A(16x4 f32) * ones(4x16) + C, chained over nB/4.
    v2f onesb; onesb.x = 1.0f; onesb.y = 1.0f;   // all-ones B: layout-agnostic
    const int nChunks = nB >> 2;                 // nB=64 -> 16 WMMAs per wave
    for (int kk = 0; kk < nChunks; ++kk) {
        const float* pr = &partials[(size_t)(ch_base + row) * nB + kk * 4 + sel * 2];
        v2f a; a.x = pr[0]; a.y = pr[1];
        acc = __builtin_amdgcn_wmma_f32_16x16x4_f32(
            /*neg_a=*/false, a, /*neg_b=*/false, onesb,
            /*c_mod=*/(short)0, acc, /*reuse_a=*/false, /*reuse_b=*/false);
    }
#else
    // Fallback: f16 WMMA (partials quantized to f16; ~2e-6 error on the mean).
    v16h onesb;
    for (int i = 0; i < 16; ++i) onesb[i] = (_Float16)1.0f;
    const int nChunks = nB >> 5;                 // nB=64 -> 2 WMMAs per wave
    for (int kk = 0; kk < nChunks; ++kk) {
        const float* pr = &partials[(size_t)(ch_base + row) * nB + kk * 32 + sel * 16];
        v16h a;
        for (int i = 0; i < 16; ++i) a[i] = (_Float16)pr[i];
        acc = __builtin_amdgcn_wmma_f32_16x16x32_f16(
            false, a, false, onesb, (short)0, acc, false, false);
    }
#endif

    // D layout (32-bit C/D 16x16): VGPR j -> M=j (lanes 0-15), M=8+j (16-31).
    // Every column holds the same row-sum; read column 0 from lanes 0 and 16.
    if (lane == 0) {
        for (int j = 0; j < 8; ++j) s_sum[ch_base + j] = acc[j];
    } else if (lane == 16) {
        for (int j = 0; j < 8; ++j) s_sum[ch_base + 8 + j] = acc[j];
    }
    __syncthreads();

    // Per-channel: mean -> round to 1e-6 (rintf = nearest-even, matches
    // jnp.round) -> searchsorted(left) -> nearest-key pick (ties -> lo).
    if (tid < C_CHANNELS) {
        const float mean = s_sum[tid] / (float)(nB * HW);
        const float q = rintf(mean * 1.0e6f) / 1.0e6f;
        int lo = 0, hi = K;
        while (lo < hi) {                      // lower_bound on sorted keys
            int mid = (lo + hi) >> 1;
            if (keys[mid] < q) lo = mid + 1; else hi = mid;
        }
        int il = lo - 1; if (il < 0) il = 0; if (il > K - 1) il = K - 1;
        int ih = lo;     if (ih > K - 1) ih = K - 1;
        const float dl = fabsf(keys[il] - q);
        const float dh = fabsf(keys[ih] - q);
        s_val[tid] = vals[(dh < dl) ? ih : il];
    }
    __syncthreads();

    // Deterministic max tree over the 256 channel values.
    for (int st = C_CHANNELS / 2; st > 0; st >>= 1) {
        if (tid < st) s_val[tid] = fmaxf(s_val[tid], s_val[tid + st]);
        __syncthreads();
    }

    const float scale = ((float)icp[0] * (float)ocp[0]) / 512.0f;  // /(32*16)
    const float v = s_val[0] * scale;
    for (int i = tid; i < outN; i += blockDim.x) out[i] = v;
}

// ---------------------------------------------------------------------------
extern "C" void kernel_launch(void* const* d_in, const int* in_sizes, int n_in,
                              void* d_out, int out_size, void* d_ws, size_t ws_size,
                              hipStream_t stream) {
    const float* x    = (const float*)d_in[0];   // layer_output (B,C,H,W) f32
    const float* keys = (const float*)d_in[1];   // sorted keys [K]
    const float* vals = (const float*)d_in[2];   // values [K]
    const int*   icp  = (const int*)d_in[3];     // in_channels (scalar)
    const int*   ocp  = (const int*)d_in[4];     // out_channels (scalar)
    float* out      = (float*)d_out;             // (56,56) f32
    float* partials = (float*)d_ws;              // [C][nB] floats (64 KB)

    const int K  = in_sizes[1];
    const int nB = in_sizes[0] / (C_CHANNELS * HW);   // 64 batches

    delay_partial_sums_kernel<<<nB * C_CHANNELS, 256, 0, stream>>>(x, partials, nB);
    delay_finalize_kernel<<<1, 2 * C_CHANNELS, 0, stream>>>(
        partials, keys, vals, icp, ocp, out, nB, K, out_size);
}
